// DistanceLayer_56848187130350
// MI455X (gfx1250) — matrix operand order repaired and checked by reading, not compiled, not run on hardware
//
#include <hip/hip_runtime.h>

// ---------------------------------------------------------------------------
// GMM weighted-Mahalanobis score for MI455X (gfx1250, wave32, WMMA + TDM).
// d[n,c] = x^T A_c x - 2 (A_c m_c)·x + m_c^T A_c m_c   (A_c = Sigma_c^-1)
// score[n] = exp2( sum_c w_c * log2 d[n,c] )
// Stage 1 GEMM per 16-sample tile: T = A_c * X^T  (64x16, K=64) via
// v_wmma_f32_16x16x32_bf16 with bf16 hi/lo split (3 products ~ fp32).
// -2*b_c folded into the WMMA accumulator init. Diagonal extracted with a
// half-wave shuffle rowdot in fp32. 256KB of pre-converted A operands are
// staged into LDS by the Tensor Data Mover (tensor_load_to_lds +
// s_wait_tensorcnt).
// ---------------------------------------------------------------------------

typedef __attribute__((ext_vector_type(16))) __bf16 v16bf;
typedef __attribute__((ext_vector_type(8)))  float  v8f;

typedef __attribute__((ext_vector_type(4))) unsigned int u32x4;
typedef __attribute__((ext_vector_type(8))) int          i32x8;
typedef __attribute__((ext_vector_type(4))) int          i32x4;

#define C_COMP 16
#define D_DIM  64
#define MAIN_WGS 256
#define WAVES_PER_WG 16
#define THREADS_MAIN (WAVES_PER_WG * 32)

#ifndef USE_TDM
#define USE_TDM 1
#endif

// workspace layout (bytes)
#define WS_A_BYTES (C_COMP * 2 /*hi,lo*/ * 4 /*mt*/ * 2 /*ks*/ * 1024)  // 262144
#define WS_B_OFF   WS_A_BYTES
#define WS_B_BYTES (C_COMP * D_DIM * 4)                                  // 4096
#define WS_CC_OFF  (WS_B_OFF + WS_B_BYTES)
#define SMEM_BYTES (WS_A_BYTES + WS_B_BYTES + 2 * C_COMP * 4)            // 266368

union OpU {
  unsigned int u[8];
  uint4 q[2];
  v16bf v;
};

static __device__ __forceinline__ unsigned short f2bf(float f) {
  unsigned u = __float_as_uint(f);
  u += 0x7fffu + ((u >> 16) & 1u);   // round-to-nearest-even
  return (unsigned short)(u >> 16);
}
static __device__ __forceinline__ float bf2f(unsigned short h) {
  return __uint_as_float(((unsigned)h) << 16);
}

// ---------------------------------------------------------------------------
// Prep: convert A_c into WMMA A-layout bf16 hi/lo operand blocks; compute
// bArr[c][k] = -2*(A_c m_c)_k and cc[c] = m_c^T A_c m_c.
// Operand block (c, split, mt, ks): 1KB = [half(2)][lane(32)][8 bf16].
// A-layout: element e of lane l -> row = 16*mt + (l%16),
//           col = 32*ks + e + 16*(l>>4).
// ---------------------------------------------------------------------------
__global__ void gmm_prep(const float* __restrict__ Ainv,
                         const float* __restrict__ means,
                         unsigned short* __restrict__ wsA,
                         float* __restrict__ wsB,
                         float* __restrict__ wsCC) {
  __shared__ float sA[D_DIM * D_DIM];
  __shared__ float sB[D_DIM];
  const int c = blockIdx.x;
  const int t = threadIdx.x;  // 256
  const float* A = Ainv + c * D_DIM * D_DIM;
  for (int i = t; i < D_DIM * D_DIM; i += 256) sA[i] = A[i];
  __syncthreads();

  if (t < D_DIM) {
    const float* m = means + c * D_DIM;
    float s = 0.f;
    for (int j = 0; j < D_DIM; ++j) s += sA[t * D_DIM + j] * m[j];
    sB[t] = s;
    wsB[c * D_DIM + t] = -2.0f * s;
  }
  __syncthreads();
  if (t == 0) {
    const float* m = means + c * D_DIM;
    float s = 0.f;
    for (int k = 0; k < D_DIM; ++k) s += m[k] * sB[k];
    wsCC[c] = s;
  }

  const int mt   = t >> 6;
  const int ks   = (t >> 5) & 1;
  const int lane = t & 31;
  const int row  = 16 * mt + (lane & 15);
  const int blkH = ((c * 2 + 0) * 4 + mt) * 2 + ks;
  const int blkL = ((c * 2 + 1) * 4 + mt) * 2 + ks;
#pragma unroll
  for (int e = 0; e < 16; ++e) {
    int col = 32 * ks + e + 16 * (lane >> 4);
    float a = sA[row * D_DIM + col];
    unsigned short hi = f2bf(a);
    unsigned short lo = f2bf(a - bf2f(hi));
    int off = (e >> 3) * 256 + lane * 8 + (e & 7);
    wsA[blkH * 512 + off] = hi;
    wsA[blkL * 512 + off] = lo;
  }
}

static __device__ __forceinline__ v16bf ldOp(const uint4* aU, int blk, int lane) {
  OpU o;
  o.q[0] = aU[blk * 64 + lane];        // half 0: elements 0..7
  o.q[1] = aU[blk * 64 + 32 + lane];   // half 1: elements 8..15
  return o.v;
}

#define WMMA(ACC, A_, B_) \
  ACC = __builtin_amdgcn_wmma_f32_16x16x32_bf16(false, (A_), false, (B_), (short)0, (ACC), false, false)

// ---------------------------------------------------------------------------
// Main kernel: 16 waves/WG, 260KB LDS (A operands + b + cc + w).
// Each wave owns one 16-sample tile at a time.
// ---------------------------------------------------------------------------
__global__ __launch_bounds__(THREADS_MAIN, 1)
void gmm_main(const float* __restrict__ X,
              const unsigned short* __restrict__ wsA,
              const float* __restrict__ wsB,
              const float* __restrict__ wsCC,
              const float* __restrict__ wts,
              float* __restrict__ out, int ntiles) {
  extern __shared__ char smem[];
  unsigned short* lA = (unsigned short*)smem;       // LDS offset 0
  float* lB  = (float*)(smem + WS_A_BYTES);
  float* lCC = (float*)(smem + WS_A_BYTES + WS_B_BYTES);
  float* lW  = lCC + C_COMP;

  const int tid = threadIdx.x;

#if USE_TDM
  // --- TDM: DMA the 256KB of A operand blocks global->LDS (one descriptor,
  //     2D tile: 8 rows x 4096 elements x 8B = 256KB, LDS offset 0).
  if (tid < 32) {  // wave 0 issues; EXEC is ignored by tensor ops
    unsigned long long ga = (unsigned long long)wsA;
    u32x4 g0;
    g0[0] = 1u;                                         // count=1, user mode
    g0[1] = 0u;                                         // lds_addr = 0
    g0[2] = (unsigned)(ga & 0xffffffffu);               // global_addr[31:0]
    g0[3] = (unsigned)((ga >> 32) & 0x01ffffffu) | 0x80000000u;  // [56:32] | type=2
    i32x8 g1;
    g1[0] = (int)(3u << 16);      // workgroup_mask=0, data_size=3 (8B)
    g1[1] = (int)(4096u << 16);   // tensor_dim0 = 4096 (low 16 -> bits 63:48)
    g1[2] = (int)(8u << 16);      // tensor_dim0 hi=0, tensor_dim1 = 8
    g1[3] = (int)(4096u << 16);   // tensor_dim1 hi=0, tile_dim0 = 4096
    g1[4] = 8;                    // tile_dim1 = 8, tile_dim2 = 0
    g1[5] = 4096;                 // tensor_dim0_stride[31:0]
    g1[6] = 0;                    // stride0 hi, tensor_dim1_stride lo
    g1[7] = 0;
    i32x4 gz4 = {0, 0, 0, 0};
    i32x8 gz8 = {0, 0, 0, 0, 0, 0, 0, 0};
    __builtin_amdgcn_tensor_load_to_lds(g0, g1, gz4, gz4, gz8, 0);
  }
  // small tables via normal loads
  for (int i = tid; i < C_COMP * D_DIM; i += THREADS_MAIN) lB[i] = wsB[i];
  if (tid < C_COMP) { lCC[tid] = wsCC[tid]; lW[tid] = wts[tid]; }
  if (tid < 32) __builtin_amdgcn_s_wait_tensorcnt(0);
#else
  {  // fallback: staged copy through VGPRs
    const uint4* src = (const uint4*)wsA;
    uint4* dst = (uint4*)lA;
    for (int i = tid; i < WS_A_BYTES / 16; i += THREADS_MAIN) dst[i] = src[i];
    for (int i = tid; i < C_COMP * D_DIM; i += THREADS_MAIN) lB[i] = wsB[i];
    if (tid < C_COMP) { lCC[tid] = wsCC[tid]; lW[tid] = wts[tid]; }
  }
#endif
  __syncthreads();

  const int lane = tid & 31;
  const int hf   = lane >> 4;  // half-wave id
  const int gwave = blockIdx.x * WAVES_PER_WG + (tid >> 5);
  const uint4* aU = (const uint4*)lA;

  for (int tile = gwave; tile < ntiles; tile += MAIN_WGS * WAVES_PER_WG) {
    // --- load X tile in B-layout: lane holds sample n = tile*16+(lane%16),
    //     features f = 16*hf + {0..15} and 16*hf + 32 + {0..15}
    const int n = tile * 16 + (lane & 15);
    const float4* p = (const float4*)(X + n * D_DIM + hf * 16);
    float xq[32];
#pragma unroll
    for (int i = 0; i < 4; ++i) {
      float4 v = p[i];
      xq[4 * i] = v.x; xq[4 * i + 1] = v.y; xq[4 * i + 2] = v.z; xq[4 * i + 3] = v.w;
    }
#pragma unroll
    for (int i = 0; i < 4; ++i) {
      float4 v = p[8 + i];
      xq[16 + 4 * i] = v.x; xq[17 + 4 * i] = v.y; xq[18 + 4 * i] = v.z; xq[19 + 4 * i] = v.w;
    }
    {  // prefetch next tile of X
      int nt = tile + MAIN_WGS * WAVES_PER_WG;
      if (nt < ntiles)
        __builtin_prefetch(X + (nt * 16 + (lane & 15)) * D_DIM + hf * 16, 0, 1);
    }

    // --- split into bf16 hi/lo B-operands (once per tile, reused for all c)
    OpU zh0u, zh1u, zl0u, zl1u;
#pragma unroll
    for (int j = 0; j < 8; ++j) {
      float a0 = xq[2 * j], a1 = xq[2 * j + 1];
      unsigned short h0 = f2bf(a0), h1 = f2bf(a1);
      zh0u.u[j] = (unsigned)h0 | ((unsigned)h1 << 16);
      unsigned short l0 = f2bf(a0 - bf2f(h0)), l1 = f2bf(a1 - bf2f(h1));
      zl0u.u[j] = (unsigned)l0 | ((unsigned)l1 << 16);
      float b0 = xq[16 + 2 * j], b1 = xq[17 + 2 * j];
      unsigned short g0 = f2bf(b0), g1 = f2bf(b1);
      zh1u.u[j] = (unsigned)g0 | ((unsigned)g1 << 16);
      unsigned short m0 = f2bf(b0 - bf2f(g0)), m1 = f2bf(b1 - bf2f(g1));
      zl1u.u[j] = (unsigned)m0 | ((unsigned)m1 << 16);
    }
    const v16bf zh0 = zh0u.v, zh1 = zh1u.v, zl0 = zl0u.v, zl1 = zl1u.v;

    float logAcc = 0.f;
#pragma unroll 1
    for (int c = 0; c < C_COMP; ++c) {
      // accumulator init = -2*b_c broadcast along columns (C-layout row k only)
      v8f acc0, acc1, acc2, acc3;
      {
        const float4* bp = (const float4*)(lB + c * D_DIM + hf * 8);
#define BINIT(ACC, MT) { float4 q0 = bp[(MT)*4]; float4 q1 = bp[(MT)*4 + 1]; \
        ACC = (v8f){q0.x, q0.y, q0.z, q0.w, q1.x, q1.y, q1.z, q1.w}; }
        BINIT(acc0, 0) BINIT(acc1, 1) BINIT(acc2, 2) BINIT(acc3, 3)
#undef BINIT
      }
      // T' = A_c * X^T - 2 b_c  (bf16x3 emulated fp32 GEMM)
#define DO_MT(ACC, MT) { \
      int bh = ((c * 2 + 0) * 4 + (MT)) * 2; \
      int bl = ((c * 2 + 1) * 4 + (MT)) * 2; \
      v16bf Ah0 = ldOp(aU, bh + 0, lane), Ah1 = ldOp(aU, bh + 1, lane); \
      v16bf Al0 = ldOp(aU, bl + 0, lane), Al1 = ldOp(aU, bl + 1, lane); \
      WMMA(ACC, Ah0, zh0); WMMA(ACC, Ah1, zh1); \
      WMMA(ACC, Al0, zh0); WMMA(ACC, Al1, zh1); \
      WMMA(ACC, Ah0, zl0); WMMA(ACC, Ah1, zl1); }
      DO_MT(acc0, 0) DO_MT(acc1, 1) DO_MT(acc2, 2) DO_MT(acc3, 3)
#undef DO_MT

      // rowdot: d = sum_k x[n,k]*T'[k,n] ; lane-local pairs + xor-16 pairs
      float sum = 0.f;
#pragma unroll
      for (int j = 0; j < 8; ++j) {
        float aA = hf ? acc1[j] : acc0[j];
        float xA = hf ? xq[8 + j] : xq[j];
        float aB = hf ? acc3[j] : acc2[j];
        float xB = hf ? xq[24 + j] : xq[16 + j];
        sum += xA * aA + xB * aB;
      }
#pragma unroll
      for (int j = 0; j < 16; ++j) {
        const int jm = j & 7;
        float pass = hf ? ((j < 8) ? xq[j] : xq[j + 8])
                        : ((j < 8) ? xq[8 + j] : xq[j + 16]);
        float xs = __shfl_xor(pass, 16, 32);
        float av;
        if (j < 8) av = hf ? acc0[jm] : acc1[jm];
        else       av = hf ? acc2[jm] : acc3[jm];
        sum += xs * av;
      }
      sum += __shfl_xor(sum, 16, 32);   // combine half-waves: full k range
      float dd = sum + lCC[c];
      dd = fmaxf(dd, 1e-30f);
      // v_log_f32 is log2 on AMDGPU; v_exp_f32 is exp2
      logAcc += lW[c] * __builtin_amdgcn_logf(dd);
    }
    float score = __builtin_amdgcn_exp2f(logAcc);
    if (hf == 0) out[tile * 16 + lane] = score;
  }
}

extern "C" void kernel_launch(void* const* d_in, const int* in_sizes, int n_in,
                              void* d_out, int out_size, void* d_ws, size_t ws_size,
                              hipStream_t stream) {
  const float* X     = (const float*)d_in[0];
  const float* Ainv  = (const float*)d_in[1];
  const float* means = (const float*)d_in[2];
  const float* wts   = (const float*)d_in[3];
  float* out = (float*)d_out;
  (void)n_in; (void)out_size; (void)ws_size;

  const int N = in_sizes[0] / D_DIM;
  const int ntiles = N / 16;

  unsigned short* wsA = (unsigned short*)d_ws;
  float* wsB  = (float*)((char*)d_ws + WS_B_OFF);
  float* wsCC = (float*)((char*)d_ws + WS_CC_OFF);

  hipLaunchKernelGGL(gmm_prep, dim3(C_COMP), dim3(256), 0, stream,
                     Ainv, means, wsA, wsB, wsCC);
  hipLaunchKernelGGL(gmm_main, dim3(MAIN_WGS), dim3(THREADS_MAIN), SMEM_BYTES, stream,
                     X, wsA, wsB, wsCC, wts, out, ntiles);
}
